// CachedReadoutModel_27504970564009
// MI455X (gfx1250) — matrix operand order repaired and checked by reading, not compiled, not run on hardware
//
#include <hip/hip_runtime.h>
#include <hip/hip_bf16.h>

#define IN_F   192
#define NCOMP  128
#define HIDDEN 128
#define NSPEC  10
#define KTILES 6        // 192 / 32
#define NTILES 8        // 128 / 16
#define BTILES (KTILES * NTILES)   // 48 B-operand tiles
#define BWORDS 12288    // packed B table: 48*32*8 dwords = 48 KB
#define TPW    4        // atom-tiles per wave per block
#define BPIPE  3        // B tiles prefetched ahead (4 rotating buffers)

typedef __attribute__((ext_vector_type(16))) __bf16   v16bf;
typedef __attribute__((ext_vector_type(8)))  float    v8f;
typedef __attribute__((ext_vector_type(4)))  unsigned v4u;
typedef __attribute__((ext_vector_type(8)))  unsigned v8u;

union V16BF { unsigned int u[8]; v16bf v; };

// f32x2 -> packed bf16 (RNE) in one VALU op.
__device__ __forceinline__ unsigned int cvt_pk_bf16(float lo, float hi) {
#if __has_builtin(__builtin_amdgcn_cvt_pk_bf16_f32)
    typedef __attribute__((ext_vector_type(2))) __bf16 v2bf;
    union { v2bf v; unsigned int u; } r;
    r.v = __builtin_amdgcn_cvt_pk_bf16_f32(lo, hi);
    return r.u;
#else
    unsigned int r;
    asm("v_cvt_pk_bf16_f32 %0, %1, %2" : "=v"(r) : "v"(lo), "v"(hi));
    return r;
#endif
}

// Scalar RNE pack for the prep kernels (not in the hot loop).
__device__ __forceinline__ unsigned int pack_bf16(float lo, float hi) {
    unsigned int a = __float_as_uint(lo);
    unsigned int b = __float_as_uint(hi);
    a = (a + 0x7FFFu + ((a >> 16) & 1u)) >> 16;
    b = (b + 0x7FFFu + ((b >> 16) & 1u)) >> 16;
    return (a & 0xFFFFu) | (b << 16);
}

__device__ __forceinline__ void loadB(V16BF& B, const unsigned int* Bs,
                                      int lane, int idx) {
    const uint4* bp = (const uint4*)&Bs[(idx * 32 + lane) * 8];
    uint4 b0 = bp[0], b1 = bp[1];
    B.u[0] = b0.x; B.u[1] = b0.y; B.u[2] = b0.z; B.u[3] = b0.w;
    B.u[4] = b1.x; B.u[5] = b1.y; B.u[6] = b1.z; B.u[7] = b1.w;
}

// ---------------------------------------------------------------------------
__global__ void k_zero(float* __restrict__ p, int n) {
    int i = blockIdx.x * blockDim.x + threadIdx.x;
    if (i < n) p[i] = 0.0f;
}

// ---------------------------------------------------------------------------
// Pack M = W1 @ C ([HIDDEN x IN_F]) as bf16 in the WMMA B-operand lane layout:
//   lane: column n = u*16 + (lane&15); K group = (lane>>4)*8
//   dword j holds packed K pair (kb, kb + 1), kb = kgroup + 2*(j&3) + 16*(j>>2)
// Linear index == main-kernel LDS read address: idx = ((t*8+u)*32 + lane)*8 + j.
__global__ void k_pack_B(const float* __restrict__ W1,
                         const float* __restrict__ comp,
                         unsigned int* __restrict__ Bpack) {
    int idx  = blockIdx.x * 256 + threadIdx.x;     // [0, BWORDS)
    int j    = idx & 7;
    int lane = (idx >> 3) & 31;
    int tile = idx >> 8;                           // 0..47
    int t = tile >> 3;
    int u = tile & 7;
    int n  = u * 16 + (lane & 15);
    int k0 = t * 32 + ((lane >> 4) * 8) + 2 * (j & 3) + 16 * (j >> 2);
    float m0 = 0.0f, m1 = 0.0f;
    const float* w = W1 + (size_t)n * NCOMP;
    for (int s = 0; s < NCOMP; ++s) {
        float ws = w[s];
        m0 += ws * comp[(size_t)s * IN_F + k0];
        m1 += ws * comp[(size_t)s * IN_F + k0 + 1];
    }
    Bpack[idx] = pack_bf16(m0, m1);
}

// ---------------------------------------------------------------------------
// Fused bias: b'[h] = b1[h] - sum_f M[h][f] * mean[f].
__global__ void k_bias(const float* __restrict__ W1,
                       const float* __restrict__ comp,
                       const float* __restrict__ b1,
                       const float* __restrict__ mean,
                       float* __restrict__ bias2) {
    int h = threadIdx.x;
    float acc = b1[h];
    const float* w = W1 + (size_t)h * NCOMP;
    for (int f = 0; f < IN_F; ++f) {
        float m = 0.0f;
        for (int s = 0; s < NCOMP; ++s) m += w[s] * comp[(size_t)s * IN_F + f];
        acc -= m * mean[f];
    }
    bias2[h] = acc;
}

// ---------------------------------------------------------------------------
// Main fused kernel. Per block: TDM-stage the 48 KB packed weight table into
// LDS once, then 8 waves each process TPW tiles of 16 atoms:
//   h_pre = x @ M^T + b'  via v_wmma_f32_16x16x32_bf16, B software-pipelined
//   from LDS through 4 rotating register buffers (loads issued BPIPE tiles
//   ahead so the DS latency is hidden behind WMMA issue).
__global__ void __launch_bounds__(256, 1)
k_main(const float* __restrict__ x,
       const float* __restrict__ node_attrs,
       const int*   __restrict__ batch_map,
       const unsigned int* __restrict__ Bpack,
       const float* __restrict__ bias2,
       const float* __restrict__ w2,
       const float* __restrict__ b2,
       const float* __restrict__ shifts,
       float* __restrict__ delta,
       int nTiles) {
    __shared__ unsigned int Bs[BWORDS];   // 48 KB staged weight table
    __shared__ float red[8 * 16];

    int lane = threadIdx.x & 31;
    int wave = threadIdx.x >> 5;

    // --- TDM: one wave DMAs Bpack (global) -> Bs (LDS), 1D tile of 12288 u32.
    if (wave == 0) {
        unsigned long long ga = (unsigned long long)(uintptr_t)Bpack;
        unsigned lds_base = (unsigned)(uintptr_t)(&Bs[0]);
        v4u g0;
        g0[0] = 0x1u;                                    // count=1, user mode
        g0[1] = lds_base;                                // lds_addr (bytes)
        g0[2] = (unsigned)(ga & 0xFFFFFFFFu);            // global_addr[31:0]
        g0[3] = (unsigned)((ga >> 32) & 0x01FFFFFFu)     // global_addr[56:32]
                | 0x80000000u;                           // type=2 ("image")
        v8u g1;
        g1[0] = 0x00020000u;                             // data_size=4B
        g1[1] = (BWORDS & 0xFFFFu) << 16;                // tensor_dim0[15:0]
        g1[2] = ((BWORDS >> 16) & 0xFFFFu) | (1u << 16); // dim0 hi | tensor_dim1=1
        g1[3] = (BWORDS & 0xFFFFu) << 16;                // tile_dim0 = BWORDS
        g1[4] = 1u;                                      // tile_dim1 = 1
        g1[5] = (unsigned)BWORDS;                        // tensor_dim0_stride lo
        g1[6] = 0u;
        g1[7] = 0u;
        asm volatile("tensor_load_to_lds %0, %1" :: "s"(g0), "s"(g1) : "memory");
        __builtin_amdgcn_s_wait_tensorcnt(0);
        asm volatile("" ::: "memory");
    }
    __syncthreads();

    int m     = lane & 15;
    int khalf = (lane >> 4) * 8;

#pragma unroll 1
    for (int it = 0; it < TPW; ++it) {
        int tileId = (blockIdx.x * TPW + it) * 8 + wave;
        if (tileId >= nTiles) continue;                 // wave-uniform guard
        int atomBase = tileId * 16;

        // A layout (16-bit A 16x32): lane l -> row M = l&15,
        // K groups (l>>4)*8 .. +7 (dwords 0-3) and +16 .. +23 (dwords 4-7).
        const float* rowp = x + (size_t)(atomBase + m) * IN_F + khalf;

        v8f acc[NTILES] = {};
        V16BF Bb[4];
        loadB(Bb[0], Bs, lane, 0);          // prime the 3-deep B pipeline
        loadB(Bb[1], Bs, lane, 1);
        loadB(Bb[2], Bs, lane, 2);

#pragma unroll
        for (int t = 0; t < KTILES; ++t) {
            const float4* gp1 = (const float4*)(rowp + t * 32);
            const float4* gp2 = (const float4*)(rowp + t * 32 + 16);
            float4 a0 = gp1[0], a1 = gp1[1], a2 = gp2[0], a3 = gp2[1];
            V16BF A;
            A.u[0] = cvt_pk_bf16(a0.x, a0.y); A.u[1] = cvt_pk_bf16(a0.z, a0.w);
            A.u[2] = cvt_pk_bf16(a1.x, a1.y); A.u[3] = cvt_pk_bf16(a1.z, a1.w);
            A.u[4] = cvt_pk_bf16(a2.x, a2.y); A.u[5] = cvt_pk_bf16(a2.z, a2.w);
            A.u[6] = cvt_pk_bf16(a3.x, a3.y); A.u[7] = cvt_pk_bf16(a3.z, a3.w);
#pragma unroll
            for (int u = 0; u < NTILES; ++u) {
                int idx = t * NTILES + u;
                int nxt = idx + BPIPE;
                if (nxt < BTILES) loadB(Bb[nxt & 3], Bs, lane, nxt);
                acc[u] = __builtin_amdgcn_wmma_f32_16x16x32_bf16(
                    false, A.v, false, Bb[idx & 3].v, (short)0, acc[u],
                    false, false);
            }
        }

        // Epilogue. C layout: lane l, vgpr v -> row M = v + 8*(l>=16), N = l&15.
        float partial[8];
#pragma unroll
        for (int v = 0; v < 8; ++v) partial[v] = 0.0f;
#pragma unroll
        for (int u = 0; u < NTILES; ++u) {
            int h = u * 16 + (lane & 15);
            float wv = w2[h];
            float bb = bias2[h];
#pragma unroll
            for (int v = 0; v < 8; ++v) {
                float z = acc[u][v] + bb;
                float s = z / (1.0f + __expf(-z));      // SiLU
                partial[v] += wv * s;
            }
        }
#pragma unroll
        for (int v = 0; v < 8; ++v) {
            float p = partial[v];
            p += __shfl_xor(p, 1, 32);
            p += __shfl_xor(p, 2, 32);
            p += __shfl_xor(p, 4, 32);
            p += __shfl_xor(p, 8, 32);
            partial[v] = p;
        }
        if ((lane & 15) == 0) {
            int half = lane >> 4;
#pragma unroll
            for (int v = 0; v < 8; ++v) red[wave * 16 + half * 8 + v] = partial[v];
        }
        asm volatile("s_wait_dscnt 0" ::: "memory");    // same-wave DS ordering

        if (lane < 16) {
            int atom = atomBase + lane;
            float val = red[wave * 16 + lane] + b2[0];
            const float* na = node_attrs + (size_t)atom * NSPEC;
            float best = na[0]; int bi = 0;
#pragma unroll
            for (int s2 = 1; s2 < NSPEC; ++s2) {
                float vv = na[s2];
                if (vv > best) { best = vv; bi = s2; }  // first-max == argmax
            }
            val += shifts[bi];
            atomicAdd(&delta[batch_map[atom]], val);
        }
    }
}

// ---------------------------------------------------------------------------
__global__ void k_final(const float* __restrict__ base,
                        const float* __restrict__ delta,
                        float* __restrict__ out, int n) {
    int i = blockIdx.x * blockDim.x + threadIdx.x;
    if (i < n) out[i] = base[i] + delta[i];
}

// ---------------------------------------------------------------------------
extern "C" void kernel_launch(void* const* d_in, const int* in_sizes, int n_in,
                              void* d_out, int out_size, void* d_ws, size_t ws_size,
                              hipStream_t stream) {
    const float* x           = (const float*)d_in[0];
    const float* node_attrs  = (const float*)d_in[1];
    const int*   batch_map   = (const int*)  d_in[2];
    const float* base_energy = (const float*)d_in[3];
    const float* pca_mean    = (const float*)d_in[4];
    const float* pca_comp    = (const float*)d_in[5];
    const float* W1          = (const float*)d_in[6];
    const float* b1          = (const float*)d_in[7];
    const float* W2          = (const float*)d_in[8];
    const float* b2          = (const float*)d_in[9];
    const float* shifts      = (const float*)d_in[10];

    int nAtoms  = in_sizes[0] / IN_F;     // 1,000,000 (divisible by 16)
    int nGraphs = out_size / 2;           // 16384

    unsigned int* Bpack = (unsigned int*)d_ws;                       // 48 KB
    float* bias2 = (float*)((char*)d_ws + BWORDS * sizeof(unsigned));

    float* outFinal = (float*)d_out;
    float* outDelta = outFinal + nGraphs;

    k_zero  <<<(nGraphs + 255) / 256, 256, 0, stream>>>(outDelta, nGraphs);
    k_pack_B<<<BWORDS / 256, 256, 0, stream>>>(W1, pca_comp, Bpack);
    k_bias  <<<1, 128, 0, stream>>>(W1, pca_comp, b1, pca_mean, bias2);

    int nTiles  = nAtoms / 16;
    int nBlocks = (nTiles + 8 * TPW - 1) / (8 * TPW);
    k_main<<<nBlocks, 256, 0, stream>>>(x, node_attrs, batch_map, Bpack, bias2,
                                        W2, b2, shifts, outDelta, nTiles);

    k_final<<<(nGraphs + 255) / 256, 256, 0, stream>>>(base_energy, outDelta,
                                                       outFinal, nGraphs);
}
